// SlidingWindowCrossAttention_9294309228766
// MI455X (gfx1250) — matrix operand order repaired and checked by reading, not compiled, not run on hardware
//
#include <hip/hip_runtime.h>

// ---------------- problem constants ----------------
#define NB     4
#define DIM    256
#define Hh     128
#define Ww     128
#define HW     (Hh*Ww)        // 16384
#define AD     64
#define HEADS  4
#define HD     16             // head dim

typedef __attribute__((ext_vector_type(16))) __bf16 v16bf;
typedef __attribute__((ext_vector_type(8)))  __bf16 v8bf;
typedef __attribute__((ext_vector_type(8)))  float  v8f;

// native fp32 -> bf16 (backend selects v_cvt_(pk_)bf16_f32 on gfx1250)
static __device__ __forceinline__ __bf16 f2bf(float f) { return (__bf16)f; }

// reflect padding map for p in [-2, 129] (jnp 'reflect': -1->1, 128->126)
static __device__ __forceinline__ int refl(int p) {
    p = (p < 0) ? -p : p;
    p = (p > 127) ? (254 - p) : p;
    return p;
}

#define WMMA_BF16(A, B, C) \
    __builtin_amdgcn_wmma_f32_16x16x32_bf16(false, (A), false, (B), (short)0, (C), false, false)

// =====================================================================
// Kernel P: convert weights fp32->bf16, pre-swizzled into WMMA A-layout.
//   qkv weights (64x256): wsw[((t*8 + c)*32 + lane)*16 + i]
//       lane=(m,half): element i -> W[16t+m][c*32 + 16*half + i]
//   Wo (256x64):        wos[((t*2 + c)*32 + lane)*16 + i], t=0..15, c=0..1
// =====================================================================
__global__ __launch_bounds__(256)
void prep_weights_kernel(const float* __restrict__ Wq, const float* __restrict__ Wk,
                         const float* __restrict__ Wv, const float* __restrict__ Wo,
                         __bf16* __restrict__ wqs, __bf16* __restrict__ wks,
                         __bf16* __restrict__ wvs, __bf16* __restrict__ wos)
{
    const int j = blockIdx.x * 256 + threadIdx.x;    // 0..16383
    const int i    = j & 15;
    const int lane = (j >> 4) & 31;
    const int m    = lane & 15;
    const int half = lane >> 4;
    const int koff = 16 * half + i;
    {   // qkv weights: c = 0..7, t = 0..3
        const int c = (j >> 9) & 7;
        const int t = j >> 12;
        const int src = (t * 16 + m) * DIM + c * 32 + koff;
        wqs[j] = f2bf(Wq[src]);
        wks[j] = f2bf(Wk[src]);
        wvs[j] = f2bf(Wv[src]);
    }
    {   // Wo: c = 0..1, t = 0..15
        const int c = (j >> 9) & 1;
        const int t = j >> 10;
        const int src = (t * 16 + m) * AD + c * 32 + koff;
        wos[j] = f2bf(Wo[src]);
    }
}

// =====================================================================
// Kernel A: q/k/v projections (bf16 WMMA), L2-norm of q and k.
// One wave = 16 pixels of one batch. Output layout [b][head][pixel][hd] bf16.
// =====================================================================
__global__ __launch_bounds__(256)
void proj_qkv_kernel(const float* __restrict__ qin, const float* __restrict__ kvin,
                     const __bf16* __restrict__ wqs, const __bf16* __restrict__ wks,
                     const __bf16* __restrict__ wvs,
                     __bf16* __restrict__ qn, __bf16* __restrict__ kn,
                     __bf16* __restrict__ vn)
{
    const int lane = threadIdx.x & 31;
    const int wv   = threadIdx.x >> 5;
    const int tile = blockIdx.x * 8 + wv;       // 4096 tiles total
    const int b    = tile >> 10;                // 1024 tiles per batch
    const int p0   = (tile & 1023) << 4;        // pixel base
    const int m    = lane & 15;
    const int half = lane >> 4;

    const float* xq  = qin  + (size_t)b * DIM * HW;
    const float* xkv = kvin + (size_t)b * DIM * HW;

    v8f aq[HEADS], ak[HEADS], av[HEADS];
#pragma unroll
    for (int t = 0; t < HEADS; ++t)
#pragma unroll
        for (int r = 0; r < 8; ++r) { aq[t][r] = 0.f; ak[t][r] = 0.f; av[t][r] = 0.f; }

#pragma unroll
    for (int c = 0; c < 8; ++c) {               // K chunks of 32 (DIM=256)
        // B tiles: K = c*32 + 16*half + i (B-layout), N = pixel = m
        v16bf bq, bkv;
        const int kb = c * 32 + 16 * half;
#pragma unroll
        for (int i = 0; i < 16; ++i) {
            bq[i]  = f2bf(xq [(size_t)(kb + i) * HW + p0 + m]);
            bkv[i] = f2bf(xkv[(size_t)(kb + i) * HW + p0 + m]);
        }
#pragma unroll
        for (int t = 0; t < HEADS; ++t) {
            const size_t aoff = (size_t)((t * 8 + c) * 32 + lane) * 16;
            v16bf aw;
            aw = *(const v16bf*)(wqs + aoff);
            aq[t] = WMMA_BF16(aw, bq, aq[t]);
            aw = *(const v16bf*)(wks + aoff);
            ak[t] = WMMA_BF16(aw, bkv, ak[t]);
            aw = *(const v16bf*)(wvs + aoff);
            av[t] = WMMA_BF16(aw, bkv, av[t]);
        }
    }

    // D layout: lane holds (pixel = m), hd = r + 8*half within head t.
#pragma unroll
    for (int t = 0; t < HEADS; ++t) {
        const size_t base = (((size_t)(b * HEADS + t)) * HW + p0 + m) * HD + half * 8;

        float ssq = 0.f;
#pragma unroll
        for (int r = 0; r < 8; ++r) ssq += aq[t][r] * aq[t][r];
        ssq += __shfl_xor(ssq, 16, 32);
        float scq = 1.f / fmaxf(sqrtf(ssq), 1e-12f);
        v8bf oq;
#pragma unroll
        for (int r = 0; r < 8; ++r) oq[r] = f2bf(aq[t][r] * scq);
        *(v8bf*)(qn + base) = oq;

        float ssk = 0.f;
#pragma unroll
        for (int r = 0; r < 8; ++r) ssk += ak[t][r] * ak[t][r];
        ssk += __shfl_xor(ssk, 16, 32);
        float sck = 1.f / fmaxf(sqrtf(ssk), 1e-12f);
        v8bf ok;
#pragma unroll
        for (int r = 0; r < 8; ++r) ok[r] = f2bf(ak[t][r] * sck);
        *(v8bf*)(kn + base) = ok;

        v8bf ov;
#pragma unroll
        for (int r = 0; r < 8; ++r) ov[r] = f2bf(av[t][r]);
        *(v8bf*)(vn + base) = ov;
    }
}

// =====================================================================
// Kernel B: windowed cosine attention. One wave = one (b, head, 4x4 query tile).
// Key/value union tile = 8x8 (reflect-padded). attn^T via WMMA, lane-local
// softmax (+1 shfl), then p x V via WMMA with V staged in LDS.
// =====================================================================
__global__ __launch_bounds__(256)
void attn_kernel(const __bf16* __restrict__ qn, const __bf16* __restrict__ kn,
                 const __bf16* __restrict__ vn, __bf16* __restrict__ ao)
{
    __shared__ __bf16 lvs[8][64 * HD];          // 16 KB: per-wave V union tile
    const int lane = threadIdx.x & 31;
    const int wv   = threadIdx.x >> 5;
    __bf16* lv = &lvs[wv][0];

    const int g   = blockIdx.x * 8 + wv;        // 16384 tiles total
    const int b   = g >> 12;                    // 4096 per batch
    const int rem = g & 4095;
    const int n   = rem >> 10;                  // head
    const int t   = rem & 1023;                 // spatial tile (32x32 grid)
    const int ty0 = (t >> 5) << 2;
    const int tx0 = (t & 31) << 2;
    const int m    = lane & 15;
    const int half = lane >> 4;
    const size_t hb = ((size_t)(b * HEADS + n)) * HW;

    // Stage V union into LDS: 64 pixels x 16 hd (bf16). 32 B per lane per pixel.
#pragma unroll
    for (int w = lane; w < 64; w += 32) {
        const int sy = refl(ty0 - 2 + (w >> 3));
        const int sx = refl(tx0 - 2 + (w & 7));
        const v8bf* src = (const v8bf*)(vn + (hb + sy * Ww + sx) * HD);
        *(v8bf*)(lv + w * HD)     = src[0];
        *(v8bf*)(lv + w * HD + 8) = src[1];
    }

    // q^T as B-matrix: N = query = m, K = hd (16..31 zero-padded -> elements 8..15 zero)
    const int qy = ty0 + (m >> 2), qx = tx0 + (m & 3);
    v16bf qB;
    {
        v8bf q0 = *(const v8bf*)(qn + (hb + qy * Ww + qx) * HD + half * 8);
        const __bf16 z = f2bf(0.f);
#pragma unroll
        for (int j = 0; j < 8; ++j) {
            qB[j]     = q0[j];
            qB[8 + j] = z;
        }
    }

    // GEMM1: attn^T, 4 key groups of 16. A = K-union (M=key, K=hd padded).
    v8f pT[4];
#pragma unroll
    for (int gk = 0; gk < 4; ++gk) {
        const int w  = gk * 16 + m;
        const int sy = refl(ty0 - 2 + (w >> 3));
        const int sx = refl(tx0 - 2 + (w & 7));
        v8bf kv = *(const v8bf*)(kn + (hb + sy * Ww + sx) * HD + half * 8);
        v16bf kA;
        const __bf16 z = f2bf(0.f);
#pragma unroll
        for (int j = 0; j < 8; ++j) { kA[j] = kv[j]; kA[8 + j] = z; }
        v8f zero;
#pragma unroll
        for (int r = 0; r < 8; ++r) zero[r] = 0.f;
        pT[gk] = WMMA_BF16(kA, qB, zero);
    }

    // Lane-local softmax: lane holds scores for query m, keys w = 16*gk + 8*half + r.
    const int tyq = m >> 2, txq = m & 3;
    float pv[4][8];
    float mx = -1e30f;
#pragma unroll
    for (int gk = 0; gk < 4; ++gk)
#pragma unroll
        for (int r = 0; r < 8; ++r) {
            const int w  = gk * 16 + half * 8 + r;
            const int wy = w >> 3, wx = w & 7;
            const bool ok = (wy >= tyq) & (wy <= tyq + 4) & (wx >= txq) & (wx <= txq + 4);
            const float val = ok ? pT[gk][r] : -1e30f;
            pv[gk][r] = val;
            mx = fmaxf(mx, val);
        }
    mx = fmaxf(mx, __shfl_xor(mx, 16, 32));
    float sum = 0.f;
#pragma unroll
    for (int gk = 0; gk < 4; ++gk)
#pragma unroll
        for (int r = 0; r < 8; ++r) {
            const float e = __expf(pv[gk][r] - mx);   // masked entries underflow to 0
            pv[gk][r] = e;
            sum += e;
        }
    sum += __shfl_xor(sum, 16, 32);
    const float inv = 1.f / sum;

    // Repack attn^T D-layout directly into A-layout of GEMM2 (lane-local).
    v16bf pA[2];
#pragma unroll
    for (int c = 0; c < 2; ++c)
#pragma unroll
        for (int j = 0; j < 8; ++j) {
            pA[c][j]     = f2bf(pv[2 * c][j] * inv);
            pA[c][8 + j] = f2bf(pv[2 * c + 1][j] * inv);
        }

    asm volatile("s_wait_dscnt 0" ::: "memory");      // V staging visible to ds reads

    // GEMM2: out = p x V.  B from LDS: K = 32c + 8*half + (i<8 ? i : 8+i), N = hd = m.
    v8f ot;
#pragma unroll
    for (int r = 0; r < 8; ++r) ot[r] = 0.f;
#pragma unroll
    for (int c = 0; c < 2; ++c) {
        v16bf vB;
#pragma unroll
        for (int i = 0; i < 16; ++i)
            vB[i] = lv[(32 * c + 8 * half + (i < 8 ? i : 8 + i)) * HD + m];
        ot = WMMA_BF16(pA[c], vB, ot);
    }

    // D: M = query local = r + 8*half, N = hd = m.
#pragma unroll
    for (int r = 0; r < 8; ++r) {
        const int ql = r + 8 * half;
        const int oy = ty0 + (ql >> 2), ox = tx0 + (ql & 3);
        ao[(hb + oy * Ww + ox) * HD + m] = f2bf(ot[r]);
    }
}

// =====================================================================
// Kernel C: output projection out = Wo (256x64) x attn_out. One wave = 16 pixels.
// =====================================================================
__global__ __launch_bounds__(256)
void outproj_kernel(const __bf16* __restrict__ ao, const __bf16* __restrict__ wos,
                    float* __restrict__ out)
{
    const int lane = threadIdx.x & 31;
    const int wv   = threadIdx.x >> 5;
    const int tile = blockIdx.x * 8 + wv;       // 4096 tiles
    const int b    = tile >> 10;
    const int p0   = (tile & 1023) << 4;
    const int m    = lane & 15;
    const int half = lane >> 4;

    // B tiles: chunk c covers channels K = 32c + 16*half + i -> head 2c+half, hd=i.
    v16bf bT[2];
#pragma unroll
    for (int c = 0; c < 2; ++c) {
        const int head = 2 * c + half;
        const v8bf* src = (const v8bf*)(ao + (((size_t)(b * HEADS + head)) * HW + p0 + m) * HD);
        v8bf lo = src[0], hi = src[1];
#pragma unroll
        for (int j = 0; j < 8; ++j) { bT[c][j] = lo[j]; bT[c][8 + j] = hi[j]; }
    }

#pragma unroll 4
    for (int t = 0; t < 16; ++t) {              // 16 M-tiles of 16 out channels
        v8f acc;
#pragma unroll
        for (int r = 0; r < 8; ++r) acc[r] = 0.f;
#pragma unroll
        for (int c = 0; c < 2; ++c) {
            v16bf aw = *(const v16bf*)(wos + (size_t)((t * 2 + c) * 32 + lane) * 16);
            acc = WMMA_BF16(aw, bT[c], acc);
        }
#pragma unroll
        for (int r = 0; r < 8; ++r)
            out[((size_t)(b * DIM + t * 16 + r + 8 * half)) * HW + p0 + m] = acc[r];
    }
}

// =====================================================================
extern "C" void kernel_launch(void* const* d_in, const int* in_sizes, int n_in,
                              void* d_out, int out_size, void* d_ws, size_t ws_size,
                              hipStream_t stream)
{
    (void)in_sizes; (void)n_in; (void)out_size; (void)ws_size;
    const float* qin  = (const float*)d_in[0];
    const float* kvin = (const float*)d_in[1];
    const float* Wq   = (const float*)d_in[2];
    const float* Wk   = (const float*)d_in[3];
    const float* Wv   = (const float*)d_in[4];
    const float* Wo   = (const float*)d_in[5];
    float* out = (float*)d_out;

    __bf16* ws = (__bf16*)d_ws;
    const size_t T = (size_t)NB * HEADS * HW * HD;   // 4,194,304 elems = 8 MB each
    __bf16* qn  = ws;
    __bf16* kn  = ws + T;
    __bf16* vn  = ws + 2 * T;
    __bf16* ao  = ws + 3 * T;
    __bf16* wqs = ws + 4 * T;                        // 16384 elems each
    __bf16* wks = wqs + 16384;
    __bf16* wvs = wks + 16384;
    __bf16* wos = wvs + 16384;

    prep_weights_kernel<<<64, 256, 0, stream>>>(Wq, Wk, Wv, Wo, wqs, wks, wvs, wos);
    proj_qkv_kernel<<<512, 256, 0, stream>>>(qin, kvin, wqs, wks, wvs, qn, kn, vn);
    attn_kernel    <<<2048, 256, 0, stream>>>(qn, kn, vn, ao);
    outproj_kernel <<<512, 256, 0, stream>>>(ao, wos, out);
}
